// TimeMoeSparseExpertsLayer_49469433315533
// MI455X (gfx1250) — matrix operand order repaired and checked by reading, not compiled, not run on hardware
//
#include <hip/hip_runtime.h>
#include <hip/hip_bf16.h>
#include <math.h>

// ---------------------------------------------------------------------------
// Types
// ---------------------------------------------------------------------------
typedef __attribute__((ext_vector_type(16))) __bf16 v16bf;
typedef __attribute__((ext_vector_type(8)))  float  v8f;
typedef __attribute__((ext_vector_type(4)))  unsigned int u32x4;
typedef __attribute__((ext_vector_type(4)))  float  f32x4;
typedef __attribute__((ext_vector_type(2)))  unsigned int u32x2;

union Frag {
    u32x4 q[2];
    v16bf v;
};

// Problem constants
#define T_TOTAL   32768   // 4 * 8192 tokens
#define DMODEL    512
#define I_MOE     1024
#define I_SH      2048
#define NEXPERT   8
#define TCHUNK    8192    // tokens per h-buffer chunk
#define BM        128     // M tile (tokens) per block
#define BN        64      // N tile per block
#define BK        32      // K step

__device__ __forceinline__ unsigned short f2bf(float f) {
    unsigned int u = __float_as_uint(f);
    unsigned int r = u + 0x7FFFu + ((u >> 16) & 1u);   // round-to-nearest-even
    return (unsigned short)(r >> 16);
}

// ---------------------------------------------------------------------------
// Zero f32 buffer (float4 granularity, exact grid)
// ---------------------------------------------------------------------------
__global__ __launch_bounds__(256) void zero_f32_kernel(float* __restrict__ p) {
    size_t i = (size_t)blockIdx.x * blockDim.x + threadIdx.x;
    ((f32x4*)p)[i] = (f32x4){0.f, 0.f, 0.f, 0.f};
}

// ---------------------------------------------------------------------------
// Cast x (f32) -> bf16, 4 elements per thread, exact grid
// ---------------------------------------------------------------------------
__global__ __launch_bounds__(256) void cast_x_kernel(const float* __restrict__ x,
                                                     unsigned short* __restrict__ xb) {
    size_t i = (size_t)blockIdx.x * blockDim.x + threadIdx.x;
    f32x4 v = ((const f32x4*)x)[i];
    u32x2 o;
    o.x = (unsigned int)f2bf(v.x) | ((unsigned int)f2bf(v.y) << 16);
    o.y = (unsigned int)f2bf(v.z) | ((unsigned int)f2bf(v.w) << 16);
    ((u32x2*)xb)[i] = o;
}

// ---------------------------------------------------------------------------
// Transpose + cast:  dst[n][k] (bf16, row stride K) = src[k][n] (f32, row
// stride ld).  Tile 32x32 via LDS, coalesced both sides.
// grid = (K/32, N/32), block = 256.
// ---------------------------------------------------------------------------
__global__ __launch_bounds__(256) void transpose_cast_kernel(const float* __restrict__ src,
                                                             int ld,
                                                             unsigned short* __restrict__ dst,
                                                             int K)
{
    __shared__ unsigned short t[32][34];
    const int k0 = blockIdx.x * 32;
    const int n0 = blockIdx.y * 32;
    const int r  = threadIdx.x >> 3;         // 0..31
    const int c4 = (threadIdx.x & 7) * 4;    // 0,4,..,28

    const f32x4 v = *(const f32x4*)(src + (size_t)(k0 + r) * ld + n0 + c4);
    t[r][c4 + 0] = f2bf(v.x);
    t[r][c4 + 1] = f2bf(v.y);
    t[r][c4 + 2] = f2bf(v.z);
    t[r][c4 + 3] = f2bf(v.w);
    __syncthreads();

    // write transposed: row n = r, k chunk = c4
    u32x2 o;
    o.x = (unsigned int)t[c4 + 0][r] | ((unsigned int)t[c4 + 1][r] << 16);
    o.y = (unsigned int)t[c4 + 2][r] | ((unsigned int)t[c4 + 3][r] << 16);
    *(u32x2*)(dst + (size_t)(n0 + r) * K + k0 + c4) = o;
}

// ---------------------------------------------------------------------------
// Router: one wave per token.
//   logits[T,8] (raw, f32, to d_out tail), cw[T,8] = softmax prob if in top-2,
//   sgate[T] = sigmoid(x . shared_expert_gate_w)
// ---------------------------------------------------------------------------
__global__ __launch_bounds__(256) void router_kernel(const float* __restrict__ x,
                                                     const float* __restrict__ gate_w,      // [512,8]
                                                     const float* __restrict__ segw,        // [512,1]
                                                     float* __restrict__ logits_out,        // [T,8]
                                                     float* __restrict__ cw,                // [T,8]
                                                     float* __restrict__ sgate)             // [T]
{
    const int lane = threadIdx.x & 31;
    const int wave = threadIdx.x >> 5;
    const int t    = blockIdx.x * 8 + wave;

    const float* xr = x + (size_t)t * DMODEL;
    float a[9] = {0.f, 0.f, 0.f, 0.f, 0.f, 0.f, 0.f, 0.f, 0.f};

    for (int d = lane; d < DMODEL; d += 32) {
        float xv = xr[d];
#pragma unroll
        for (int e = 0; e < 8; ++e) a[e] += xv * gate_w[d * 8 + e];
        a[8] += xv * segw[d];
    }
#pragma unroll
    for (int e = 0; e < 9; ++e) {
#pragma unroll
        for (int off = 16; off > 0; off >>= 1)
            a[e] += __shfl_xor(a[e], off, 32);
    }

    if (lane == 0) {
        float mx = a[0];
#pragma unroll
        for (int e = 1; e < 8; ++e) mx = fmaxf(mx, a[e]);
        float p[8], sum = 0.f;
#pragma unroll
        for (int e = 0; e < 8; ++e) { p[e] = __expf(a[e] - mx); sum += p[e]; }
        float inv = 1.0f / sum;
        // top-2 (ties -> lowest index, matching lax.top_k)
        int i1 = 0;
#pragma unroll
        for (int e = 1; e < 8; ++e) if (p[e] > p[i1]) i1 = e;
        int i2 = (i1 == 0) ? 1 : 0;
#pragma unroll
        for (int e = 0; e < 8; ++e) if (e != i1 && p[e] > p[i2]) i2 = e;
#pragma unroll
        for (int e = 0; e < 8; ++e) {
            logits_out[(size_t)t * 8 + e] = a[e];
            float w = (e == i1) ? p[i1] * inv : ((e == i2) ? p[i2] * inv : 0.f);
            cw[(size_t)t * 8 + e] = w;
        }
        sgate[t] = 1.0f / (1.0f + __expf(-a[8]));
    }
}

// ---------------------------------------------------------------------------
// Fused gate+up GEMM:  h = silu(x @ Wg) * (x @ Wu)   -> bf16 hbuf[TCHUNK,1024]
//   A : xb  bf16 [T,512]        (rows t0 .. t0+TCHUNK)
//   B : gT/uT bf16 [1024][512]  (pre-transposed, n-major, k contiguous)
//   Block 256 thr (8 waves), tile 128x64, K-step 32, double-buffered LDS,
//   one barrier per K-step.  Wave w: ni = w&3, mi in {4*(w>>2) .. +3},
//   both gate and up accumulators -> 8 WMMA / k-step / wave.
// ---------------------------------------------------------------------------
__global__ __launch_bounds__(256) void gate_up_kernel(const unsigned short* __restrict__ xb,
                                                      const unsigned short* __restrict__ gT,
                                                      const unsigned short* __restrict__ uT,
                                                      unsigned short* __restrict__ hbuf,
                                                      int t0)
{
    __shared__ __align__(16) unsigned short lA[2][BM * BK];      // [buf][row][k]  8KB each
    __shared__ __align__(16) unsigned short lB[2][2][BN * BK];   // [buf][s][n][k] 4KB each

    const int tid  = threadIdx.x;
    const int lane = tid & 31;
    const int wave = tid >> 5;
    const int half = lane >> 4;
    const int l15  = lane & 15;

    const int nBase = blockIdx.x * BN;          // 0..I-64
    const int mLoc  = blockIdx.y * BM;          // chunk-local token base
    const int gRow  = t0 + mLoc;                // global token base

    const int ni = wave & 3;                    // n subtile
    const int mg = wave >> 2;                   // mi in {4mg .. 4mg+3}

    // staging: A 32B/thread (row = tid>>1, 16-elem chunk), B 16B/thread each
    const int arA = tid >> 1;                   // 0..127
    const int acA = (tid & 1) * 16;             // 0/16
    const int arB = tid >> 2;                   // 0..63
    const int acB = (tid & 3) * 8;              // 0/8/16/24

    const unsigned short* aSrc = xb + (size_t)(gRow + arA) * DMODEL + acA;
    const unsigned short* gSrc = gT + (size_t)(nBase + arB) * DMODEL + acB;
    const unsigned short* uSrc = uT + (size_t)(nBase + arB) * DMODEL + acB;

    v8f acc[4][2] = {};                         // [mi-idx][s]  s=0 gate, 1 up

    // prologue: stage k-step 0 into buffer 0
    {
        *(u32x4*)(&lA[0][arA * BK + acA])     = *(const u32x4*)(aSrc);
        *(u32x4*)(&lA[0][arA * BK + acA + 8]) = *(const u32x4*)(aSrc + 8);
        *(u32x4*)(&lB[0][0][arB * BK + acB])  = *(const u32x4*)(gSrc);
        *(u32x4*)(&lB[0][1][arB * BK + acB])  = *(const u32x4*)(uSrc);
    }

    int p = 0;
    for (int kt = 0; kt < DMODEL; kt += BK) {
        const bool hasNext = (kt + BK < DMODEL);
        u32x4 na0, na1, nb0, nb1;
        if (hasNext) {
            na0 = *(const u32x4*)(aSrc + kt + BK);
            na1 = *(const u32x4*)(aSrc + kt + BK + 8);
            nb0 = *(const u32x4*)(gSrc + kt + BK);
            nb1 = *(const u32x4*)(uSrc + kt + BK);
        }
        __syncthreads();                        // buf p visible; buf p^1 free
        if (hasNext) {
            *(u32x4*)(&lA[p ^ 1][arA * BK + acA])     = na0;
            *(u32x4*)(&lA[p ^ 1][arA * BK + acA + 8]) = na1;
            *(u32x4*)(&lB[p ^ 1][0][arB * BK + acB])  = nb0;
            *(u32x4*)(&lB[p ^ 1][1][arB * BK + acB])  = nb1;
        }

        // B fragments (32x16 bf16): K = half*8..+7, then 16+half*8..+7
        Frag bf[2];
#pragma unroll
        for (int s = 0; s < 2; ++s) {
            const unsigned short* bp = &lB[p][s][(ni * 16 + l15) * BK + half * 8];
            bf[s].q[0] = *(const u32x4*)(bp);
            bf[s].q[1] = *(const u32x4*)(bp + 16);
        }
        // A fragments + WMMA
#pragma unroll
        for (int mi2 = 0; mi2 < 4; ++mi2) {
            const int mi = 4 * mg + mi2;
            const unsigned short* ap = &lA[p][(mi * 16 + l15) * BK + half * 8];
            Frag af;
            af.q[0] = *(const u32x4*)(ap);        // K = half*8 .. +7
            af.q[1] = *(const u32x4*)(ap + 16);   // K = 16+half*8 .. +7
#pragma unroll
            for (int s = 0; s < 2; ++s)
                acc[mi2][s] = __builtin_amdgcn_wmma_f32_16x16x32_bf16(
                    false, af.v, false, bf[s].v, (short)0, acc[mi2][s], false, false);
        }
        p ^= 1;
    }

    // epilogue: h = silu(g) * u, fused in-register, bf16 store
#pragma unroll
    for (int mi2 = 0; mi2 < 4; ++mi2) {
        const int mi   = 4 * mg + mi2;
        const int rowB = mLoc + mi * 16 + half * 8;       // chunk-local row
        const int col  = nBase + ni * 16 + l15;
#pragma unroll
        for (int r = 0; r < 8; ++r) {
            float g  = acc[mi2][0][r];
            float u  = acc[mi2][1][r];
            float hv = (g / (1.0f + __expf(-g))) * u;
            hbuf[(size_t)(rowB + r) * I_MOE + col] = f2bf(hv);
        }
    }
}

// ---------------------------------------------------------------------------
// Down GEMM with scaled accumulate:  out[t, :] += scale[t] * (h @ Wd)[t, :]
//   A : hbuf bf16 [TCHUNK,1024]
//   B : dT   bf16 [512][1024]  (pre-transposed, n-major, k contiguous)
//   scale[t] = scale_base[t*sstride + scol]
//   Tile 128x64, double-buffered LDS, 4 WMMA / k-step / wave.
// ---------------------------------------------------------------------------
__global__ __launch_bounds__(256) void down_kernel(const unsigned short* __restrict__ hbuf,
                                                   const unsigned short* __restrict__ dT,
                                                   float* __restrict__ out,
                                                   int t0,
                                                   const float* __restrict__ scale_base,
                                                   int sstride, int scol)
{
    __shared__ __align__(16) unsigned short lA[2][BM * BK];
    __shared__ __align__(16) unsigned short lB[2][BN * BK];

    const int tid  = threadIdx.x;
    const int lane = tid & 31;
    const int wave = tid >> 5;
    const int half = lane >> 4;
    const int l15  = lane & 15;

    const int nBase = blockIdx.x * BN;          // 0..448
    const int mLoc  = blockIdx.y * BM;          // chunk-local

    const int ni = wave & 3;
    const int mg = wave >> 2;

    const int arA = tid >> 1;
    const int acA = (tid & 1) * 16;
    const int arB = tid >> 2;
    const int acB = (tid & 3) * 8;

    const unsigned short* aSrc = hbuf + (size_t)(mLoc + arA) * I_MOE + acA;
    const unsigned short* bSrc = dT + (size_t)(nBase + arB) * I_MOE + acB;

    v8f acc[4] = {};

    {
        *(u32x4*)(&lA[0][arA * BK + acA])     = *(const u32x4*)(aSrc);
        *(u32x4*)(&lA[0][arA * BK + acA + 8]) = *(const u32x4*)(aSrc + 8);
        *(u32x4*)(&lB[0][arB * BK + acB])     = *(const u32x4*)(bSrc);
    }

    int p = 0;
    for (int kt = 0; kt < I_MOE; kt += BK) {
        const bool hasNext = (kt + BK < I_MOE);
        u32x4 na0, na1, nb0;
        if (hasNext) {
            na0 = *(const u32x4*)(aSrc + kt + BK);
            na1 = *(const u32x4*)(aSrc + kt + BK + 8);
            nb0 = *(const u32x4*)(bSrc + kt + BK);
        }
        __syncthreads();
        if (hasNext) {
            *(u32x4*)(&lA[p ^ 1][arA * BK + acA])     = na0;
            *(u32x4*)(&lA[p ^ 1][arA * BK + acA + 8]) = na1;
            *(u32x4*)(&lB[p ^ 1][arB * BK + acB])     = nb0;
        }

        Frag bfr;
        {
            const unsigned short* bp = &lB[p][(ni * 16 + l15) * BK + half * 8];
            bfr.q[0] = *(const u32x4*)(bp);
            bfr.q[1] = *(const u32x4*)(bp + 16);
        }
#pragma unroll
        for (int mi2 = 0; mi2 < 4; ++mi2) {
            const int mi = 4 * mg + mi2;
            const unsigned short* ap = &lA[p][(mi * 16 + l15) * BK + half * 8];
            Frag af;
            af.q[0] = *(const u32x4*)(ap);
            af.q[1] = *(const u32x4*)(ap + 16);
            acc[mi2] = __builtin_amdgcn_wmma_f32_16x16x32_bf16(
                false, af.v, false, bfr.v, (short)0, acc[mi2], false, false);
        }
        p ^= 1;
    }

#pragma unroll
    for (int mi2 = 0; mi2 < 4; ++mi2) {
        const int mi  = 4 * mg + mi2;
        const int col = nBase + ni * 16 + l15;
#pragma unroll
        for (int r = 0; r < 8; ++r) {
            const int tok = t0 + mLoc + mi * 16 + half * 8 + r;
            const float sc = scale_base[(size_t)tok * sstride + scol];
            const size_t oi = (size_t)tok * DMODEL + col;
            out[oi] += sc * acc[mi2][r];
        }
    }
}

// ---------------------------------------------------------------------------
// Launch
// ---------------------------------------------------------------------------
extern "C" void kernel_launch(void* const* d_in, const int* in_sizes, int n_in,
                              void* d_out, int out_size, void* d_ws, size_t ws_size,
                              hipStream_t stream) {
    const float* x    = (const float*)d_in[0];   // [4,8192,512]
    const float* gw   = (const float*)d_in[1];   // [512,8]
    const float* eg   = (const float*)d_in[2];   // [8,512,1024]
    const float* eu   = (const float*)d_in[3];   // [8,512,1024]
    const float* ed   = (const float*)d_in[4];   // [8,1024,512]
    const float* sg   = (const float*)d_in[5];   // [512,2048]
    const float* su   = (const float*)d_in[6];   // [512,2048]
    const float* sd   = (const float*)d_in[7];   // [2048,512]
    const float* segw = (const float*)d_in[8];   // [512,1]

    float* out    = (float*)d_out;                         // [T,512]
    float* logits = out + (size_t)T_TOTAL * DMODEL;        // [T,8]

    // workspace layout (~52 MB total)
    char* w = (char*)d_ws;
    unsigned short* xb = (unsigned short*)w;                               // T*512 bf16 (32 MB)
    size_t off = (size_t)T_TOTAL * DMODEL * sizeof(unsigned short);
    float* cw = (float*)(w + off);                                         // T*8 f32 (1 MB)
    off += (size_t)T_TOTAL * 8 * sizeof(float);
    float* sgate = (float*)(w + off);                                      // T f32
    off += (size_t)T_TOTAL * sizeof(float);
    unsigned short* gTb = (unsigned short*)(w + off);                      // [1024][512] bf16 (1 MB)
    off += (size_t)I_MOE * DMODEL * sizeof(unsigned short);
    unsigned short* uTb = (unsigned short*)(w + off);                      // [1024][512] bf16 (1 MB)
    off += (size_t)I_MOE * DMODEL * sizeof(unsigned short);
    unsigned short* dTb = (unsigned short*)(w + off);                      // [512][1024] bf16 (1 MB)
    off += (size_t)DMODEL * I_MOE * sizeof(unsigned short);
    unsigned short* hbuf = (unsigned short*)(w + off);                     // TCHUNK*1024 bf16 (16 MB)

    // 1) zero the output accumulator region
    {
        const size_t n4 = (size_t)T_TOTAL * DMODEL / 4;
        zero_f32_kernel<<<(unsigned)(n4 / 256), 256, 0, stream>>>(out);
    }
    // 2) x -> bf16
    {
        const size_t n4 = (size_t)T_TOTAL * DMODEL / 4;
        cast_x_kernel<<<(unsigned)(n4 / 256), 256, 0, stream>>>(x, xb);
    }
    // 3) router + shared sigmoid gate
    router_kernel<<<T_TOTAL / 8, 256, 0, stream>>>(x, gw, segw, logits, cw, sgate);

    const int nChunks = T_TOTAL / TCHUNK;
    const dim3 blk(256);
    const dim3 gridGU(I_MOE / BN, TCHUNK / BM);      // 16 x 64
    const dim3 gridDN(DMODEL / BN, TCHUNK / BM);     // 8 x 64
    const dim3 gridTgu(DMODEL / 32, I_MOE / 32);     // src [512][*] -> dst [1024][512]
    const dim3 gridTdn(I_MOE / 32, DMODEL / 32);     // src [1024][512] -> dst [512][1024]

    // 4) routed experts (dense over experts, scaled by combine weight)
    for (int e = 0; e < NEXPERT; ++e) {
        transpose_cast_kernel<<<gridTgu, blk, 0, stream>>>(eg + (size_t)e * DMODEL * I_MOE, I_MOE, gTb, DMODEL);
        transpose_cast_kernel<<<gridTgu, blk, 0, stream>>>(eu + (size_t)e * DMODEL * I_MOE, I_MOE, uTb, DMODEL);
        transpose_cast_kernel<<<gridTdn, blk, 0, stream>>>(ed + (size_t)e * I_MOE * DMODEL, DMODEL, dTb, I_MOE);
        for (int c = 0; c < nChunks; ++c) {
            const int t0 = c * TCHUNK;
            gate_up_kernel<<<gridGU, blk, 0, stream>>>(xb, gTb, uTb, hbuf, t0);
            down_kernel<<<gridDN, blk, 0, stream>>>(hbuf, dTb, out, t0, cw, 8, e);
        }
    }
    // 5) shared expert (I_SH = 2048 split into two 1024-wide halves)
    for (int ic = 0; ic < 2; ++ic) {
        transpose_cast_kernel<<<gridTgu, blk, 0, stream>>>(sg + (size_t)ic * I_MOE, I_SH, gTb, DMODEL);
        transpose_cast_kernel<<<gridTgu, blk, 0, stream>>>(su + (size_t)ic * I_MOE, I_SH, uTb, DMODEL);
        transpose_cast_kernel<<<gridTdn, blk, 0, stream>>>(sd + (size_t)ic * I_MOE * DMODEL, DMODEL, dTb, I_MOE);
        for (int c = 0; c < nChunks; ++c) {
            const int t0 = c * TCHUNK;
            gate_up_kernel<<<gridGU, blk, 0, stream>>>(xb, gTb, uTb, hbuf, t0);
            down_kernel<<<gridDN, blk, 0, stream>>>(hbuf, dTb, out, t0, sgate, 1, 0);
        }
    }
    (void)in_sizes; (void)n_in; (void)out_size; (void)ws_size;
}
